// ExRestSelfAtten_77352361001094
// MI455X (gfx1250) — matrix operand order repaired and checked by reading, not compiled, not run on hardware
//
#include <hip/hip_runtime.h>
#include <hip/hip_bf16.h>

// ---------------------------------------------------------------------------
// Types for CDNA5 WMMA (wave32)
// ---------------------------------------------------------------------------
typedef __attribute__((ext_vector_type(16))) _Float16 v16h;
typedef __attribute__((ext_vector_type(8)))  float    v8f;

union HFrag { v16h v; unsigned int u[8]; };
union HPack { unsigned int u; _Float16 h[2]; };

__device__ __forceinline__ v8f zero_v8f() {
    v8f z;
#pragma unroll
    for (int i = 0; i < 8; ++i) z[i] = 0.0f;
    return z;
}

// A-fragment k order used in this file (contiguous, not AMD's table order): lane L = g*16 + m; VGPR v holds K pair starting at:
//   K = g*16 + 2*v       (p = 0/1 -> low/high half of dword)
__device__ __forceinline__ int a_koff(int v, int g) {
    return g * 16 + 2 * v;
}
// B-fragment k order used in this file (same contiguous order as A): lane L = g*16 + n; VGPR v holds K pair at K = g*16 + 2*v.

#define B_   4
#define S_   2048
#define DIN  2048
#define H_   128

// ---------------------------------------------------------------------------
// Positional-encoding table: pe[s][d] = sin/cos(s / 10000^(2*(d/2)/128)), f16
// ---------------------------------------------------------------------------
__global__ __launch_bounds__(256) void pe_kernel(_Float16* __restrict__ pe) {
    int idx = blockIdx.x * 256 + threadIdx.x;        // 2048*2048 exact
    int s = idx >> 11, d = idx & 2047;
    // inv_freq = 10000^(-2*(d/2)/128) = exp2(-(2*(d/2)) * log2(10000)/128)
    float f = exp2f(-(float)(2 * (d >> 1)) * (13.287712379549449f / 128.0f));
    float ang = (float)s * f;
    float v = (d & 1) ? cosf(ang) : sinf(ang);
    pe[idx] = (_Float16)v;
}

__global__ __launch_bounds__(256) void cvt_kernel(const float* __restrict__ src,
                                                  _Float16* __restrict__ dst, int n) {
    int i = blockIdx.x * 256 + threadIdx.x;
    if (i < n) dst[i] = (_Float16)src[i];
}

// ---------------------------------------------------------------------------
// GEMM1: h = relu(f16(x + PE) @ W1 + b1)   [8192 x 2048] @ [2048 x 128]
// 128 threads (4 waves), 64-row tile, each wave: 16 rows x 128 cols (8 WMMA accs).
// Software-pipelined: tile k+1 is loaded to registers while tile k computes.
// ---------------------------------------------------------------------------
__global__ __launch_bounds__(128) void gemm1_kernel(
    const float* __restrict__ x, const _Float16* __restrict__ pe,
    const _Float16* __restrict__ w1, const float* __restrict__ b1,
    _Float16* __restrict__ h16)
{
    constexpr int AST = 36;   // halves per A row (padded)
    constexpr int BST = 40;   // halves per B^T row (padded)
    __shared__ __align__(16) _Float16 aT[64 * AST];
    __shared__ __align__(16) _Float16 bT[128 * BST];

    const int t    = threadIdx.x;
    const int lane = t & 31;
    const int wave = t >> 5;
    const int g    = lane >> 4;
    const int ln   = lane & 15;
    const int rowBase = blockIdx.x * 64;
    const int ak   = t & 31;          // this thread's fixed A k-offset (coalesced per wave)
    const int am0  = t >> 5;          // A row base; rows am0 + 4*i

    float        xr[16];              // prefetched x tile
    _Float16     per[16];             // prefetched PE tile
    unsigned int br[16];              // prefetched W1 tile (2 halves per u32)

    // ---- prologue: load tile k0 = 0 into registers ----
#pragma unroll
    for (int i = 0; i < 16; ++i) {
        int m = i * 4 + am0;
        int row = rowBase + m;
        int s = row & (S_ - 1);
        xr[i]  = x[(size_t)row * DIN + ak];
        per[i] = pe[(size_t)s * DIN + ak];
    }
#pragma unroll
    for (int i = 0; i < 16; ++i) {
        int e = i * 128 + t;                 // 2048 u32 = 32 k-rows x 64 u32
        int k = e >> 6, n2 = (e & 63) << 1;
        br[i] = *reinterpret_cast<const unsigned int*>(&w1[(size_t)k * H_ + n2]);
    }

    v8f acc[8];
#pragma unroll
    for (int i = 0; i < 8; ++i) acc[i] = zero_v8f();

    for (int k0 = 0; k0 < DIN; k0 += 32) {
        __syncthreads();                     // previous compute done reading LDS
        // ---- commit prefetched registers to LDS ----
#pragma unroll
        for (int i = 0; i < 16; ++i) {
            int m = i * 4 + am0;
            aT[m * AST + ak] = (_Float16)(xr[i] + (float)per[i]);
        }
#pragma unroll
        for (int i = 0; i < 16; ++i) {
            int e = i * 128 + t;
            int k = e >> 6, n2 = (e & 63) << 1;
            HPack w; w.u = br[i];
            bT[n2 * BST + k]       = w.h[0];
            bT[(n2 + 1) * BST + k] = w.h[1];
        }
        __syncthreads();
        // ---- issue next tile's global loads (overlap with WMMA below) ----
        if (k0 + 32 < DIN) {
            const int kn = k0 + 32;
#pragma unroll
            for (int i = 0; i < 16; ++i) {
                int m = i * 4 + am0;
                int row = rowBase + m;
                int s = row & (S_ - 1);
                xr[i]  = x[(size_t)row * DIN + kn + ak];
                per[i] = pe[(size_t)s * DIN + kn + ak];
            }
#pragma unroll
            for (int i = 0; i < 16; ++i) {
                int e = i * 128 + t;
                int k = e >> 6, n2 = (e & 63) << 1;
                br[i] = *reinterpret_cast<const unsigned int*>(&w1[(size_t)(kn + k) * H_ + n2]);
            }
        }
        // ---- compute: A frag once; B frags batched 4 tiles at a time ----
        HFrag af;
#pragma unroll
        for (int v = 0; v < 8; ++v)
            af.u[v] = *reinterpret_cast<const unsigned int*>(
                &aT[(wave * 16 + ln) * AST + a_koff(v, g)]);
#pragma unroll
        for (int half = 0; half < 2; ++half) {
            HFrag bf[4];
#pragma unroll
            for (int j = 0; j < 4; ++j)
#pragma unroll
                for (int v = 0; v < 8; ++v)
                    bf[j].u[v] = *reinterpret_cast<const unsigned int*>(
                        &bT[((half * 4 + j) * 16 + ln) * BST + g * 16 + 2 * v]);
#pragma unroll
            for (int j = 0; j < 4; ++j)
                acc[half * 4 + j] = __builtin_amdgcn_wmma_f32_16x16x32_f16(
                    false, af.v, false, bf[j].v, (short)0, acc[half * 4 + j], false, false);
        }
    }
    // epilogue: +b1, relu, store f16. C layout: lane L: N=L%16, M=v+8*(L/16)
#pragma unroll
    for (int tt = 0; tt < 8; ++tt) {
        int n = tt * 16 + ln;
        float bn = b1[n];
#pragma unroll
        for (int v = 0; v < 8; ++v) {
            int row = rowBase + wave * 16 + v + 8 * g;
            float r = fmaxf(acc[tt][v] + bn, 0.0f);
            h16[(size_t)row * H_ + n] = (_Float16)r;
        }
    }
}

// ---------------------------------------------------------------------------
// QKV: {q,k,v}16 = h16 @ {Wq,Wk,Wv}   [8192x128] @ [128x128], blockIdx.y selects
// ---------------------------------------------------------------------------
__global__ __launch_bounds__(128) void qkv_kernel(
    const _Float16* __restrict__ h16,
    const _Float16* __restrict__ wq, const _Float16* __restrict__ wk,
    const _Float16* __restrict__ wv,
    _Float16* __restrict__ q16, _Float16* __restrict__ k16, _Float16* __restrict__ v16)
{
    constexpr int ST = 136;
    __shared__ __align__(16) _Float16 aT[64 * ST];
    __shared__ __align__(16) _Float16 bT[128 * ST];

    const _Float16* W = (blockIdx.y == 0) ? wq : (blockIdx.y == 1) ? wk : wv;
    _Float16*       O = (blockIdx.y == 0) ? q16 : (blockIdx.y == 1) ? k16 : v16;

    const int t = threadIdx.x;
    const int lane = t & 31, wave = t >> 5, g = lane >> 4, ln = lane & 15;
    const int rowBase = blockIdx.x * 64;

    // A: straight copy of 64 rows (u32 granularity)
#pragma unroll
    for (int i = 0; i < 32; ++i) {
        int e = i * 128 + t;
        int m = e >> 6, k2 = (e & 63) << 1;
        *reinterpret_cast<unsigned int*>(&aT[m * ST + k2]) =
            *reinterpret_cast<const unsigned int*>(&h16[(size_t)(rowBase + m) * H_ + k2]);
    }
    // B: full 128x128 transposed into LDS (u32 source granularity)
    for (int i = 0; i < 64; ++i) {
        int e = i * 128 + t;                 // 8192 u32
        int k = e >> 6, n2 = (e & 63) << 1;
        HPack w;
        w.u = *reinterpret_cast<const unsigned int*>(&W[(size_t)k * H_ + n2]);
        bT[n2 * ST + k]       = w.h[0];
        bT[(n2 + 1) * ST + k] = w.h[1];
    }
    __syncthreads();

    v8f acc[8];
#pragma unroll
    for (int i = 0; i < 8; ++i) acc[i] = zero_v8f();

#pragma unroll
    for (int ks = 0; ks < 4; ++ks) {
        HFrag af;
#pragma unroll
        for (int v = 0; v < 8; ++v)
            af.u[v] = *reinterpret_cast<const unsigned int*>(
                &aT[(wave * 16 + ln) * ST + ks * 32 + a_koff(v, g)]);
#pragma unroll
        for (int half = 0; half < 2; ++half) {
            HFrag bf[4];
#pragma unroll
            for (int j = 0; j < 4; ++j)
#pragma unroll
                for (int v = 0; v < 8; ++v)
                    bf[j].u[v] = *reinterpret_cast<const unsigned int*>(
                        &bT[((half * 4 + j) * 16 + ln) * ST + ks * 32 + g * 16 + 2 * v]);
#pragma unroll
            for (int j = 0; j < 4; ++j)
                acc[half * 4 + j] = __builtin_amdgcn_wmma_f32_16x16x32_f16(
                    false, af.v, false, bf[j].v, (short)0, acc[half * 4 + j], false, false);
        }
    }
#pragma unroll
    for (int tt = 0; tt < 8; ++tt)
#pragma unroll
        for (int v = 0; v < 8; ++v)
            O[(size_t)(rowBase + wave * 16 + v + 8 * g) * H_ + tt * 16 + ln] =
                (_Float16)acc[tt][v];
}

// ---------------------------------------------------------------------------
// Banded attention: one wave per 16-row block.
// scores(16x80) = Q @ Kblk^T via WMMA; softmax on the 65-wide diagonal band;
// context(16x128) = W'(16x96) @ Vblk(96x128) via WMMA; 128->2 projection.
// ---------------------------------------------------------------------------
__global__ __launch_bounds__(32) void attn_kernel(
    const _Float16* __restrict__ q16, const _Float16* __restrict__ k16,
    const _Float16* __restrict__ v16,
    const float* __restrict__ Wo, const float* __restrict__ bo,
    float* __restrict__ out, float* __restrict__ attn)
{
    constexpr int QST = 136, KST = 136, SST = 84, WST = 96, VST = 104;
    __shared__ __align__(16) _Float16 qb[16 * QST];
    __shared__ __align__(16) _Float16 kb[80 * KST];   // reused as cb (float) later
    __shared__ __align__(16) float    sb[16 * SST];
    __shared__ __align__(16) _Float16 wb[16 * WST];
    __shared__ __align__(16) _Float16 vt[128 * VST];

    const int lane = threadIdx.x;
    const int g = lane >> 4, ln = lane & 15;
    const int b  = blockIdx.x >> 7;
    const int i0 = (blockIdx.x & 127) << 4;
    const size_t base = (size_t)b * S_;
    const float rscale = 0.08838834764831845f;   // 1/sqrt(128)

    // Q rows i0..i0+15
#pragma unroll
    for (int i = 0; i < 32; ++i) {
        int e = i * 32 + lane;
        int m = e >> 6, k2 = (e & 63) << 1;
        *reinterpret_cast<unsigned int*>(&qb[m * QST + k2]) =
            *reinterpret_cast<const unsigned int*>(&q16[(base + i0 + m) * H_ + k2]);
    }
    // K band rows i0-32 .. i0+47, zero outside [0,S)
    for (int i = 0; i < 160; ++i) {
        int e = i * 32 + lane;
        int c = e >> 6, k2 = (e & 63) << 1;
        int rn = i0 - 32 + c;
        unsigned int u = 0u;
        if (rn >= 0 && rn < S_)
            u = *reinterpret_cast<const unsigned int*>(&k16[(base + rn) * H_ + k2]);
        *reinterpret_cast<unsigned int*>(&kb[c * KST + k2]) = u;
    }
    // V band, transposed to [n][c] for b32 B-fragment reads
    for (int i = 0; i < 160; ++i) {
        int e = i * 32 + lane;
        int c = e >> 6, n2 = (e & 63) << 1;
        int rn = i0 - 32 + c;
        _Float16 h0 = (_Float16)0.0f, h1 = (_Float16)0.0f;
        if (rn >= 0 && rn < S_) {
            h0 = v16[(base + rn) * H_ + n2];
            h1 = v16[(base + rn) * H_ + n2 + 1];
        }
        vt[n2 * VST + c] = h0;
        vt[(n2 + 1) * VST + c] = h1;
    }
    // zero padded K-columns 80..95 of V
    for (int i = 0; i < 64; ++i) {
        int e = i * 32 + lane;             // 128*16
        int n = e >> 4, c = 80 + (e & 15);
        vt[n * VST + c] = (_Float16)0.0f;
    }
    __syncthreads();

    // ---- scores: 16x80 = Q(16x128) @ Kblk^T(128x80) ----
    v8f accs[5];
#pragma unroll
    for (int i = 0; i < 5; ++i) accs[i] = zero_v8f();
#pragma unroll
    for (int ks = 0; ks < 4; ++ks) {
        HFrag af;
#pragma unroll
        for (int v = 0; v < 8; ++v)
            af.u[v] = *reinterpret_cast<const unsigned int*>(
                &qb[ln * QST + ks * 32 + a_koff(v, g)]);
        HFrag bf[5];
#pragma unroll
        for (int tt = 0; tt < 5; ++tt)
#pragma unroll
            for (int v = 0; v < 8; ++v)
                bf[tt].u[v] = *reinterpret_cast<const unsigned int*>(
                    &kb[(tt * 16 + ln) * KST + ks * 32 + g * 16 + 2 * v]);
#pragma unroll
        for (int tt = 0; tt < 5; ++tt)
            accs[tt] = __builtin_amdgcn_wmma_f32_16x16x32_f16(
                false, af.v, false, bf[tt].v, (short)0, accs[tt], false, false);
    }
#pragma unroll
    for (int tt = 0; tt < 5; ++tt)
#pragma unroll
        for (int v = 0; v < 8; ++v)
            sb[(v + 8 * g) * SST + tt * 16 + ln] = accs[tt][v];
    __syncthreads();

    // ---- softmax over 65-wide band; scores[m,j] = sb[m][m+64-j]/scale ----
    if (lane < 16) {
        const int m = lane;
        const int irow = i0 + m;
        float mx = -3.0e38f;
        for (int j = 0; j < 65; ++j)
            mx = fmaxf(mx, sb[m * SST + (m + 64 - j)] * rscale);
        float sum = 0.0f;
        for (int j = 0; j < 65; ++j)
            sum += __expf(sb[m * SST + (m + 64 - j)] * rscale - mx);
        const float rs = 1.0f / sum;
        for (int j = 0; j < 65; ++j) {
            float w = __expf(sb[m * SST + (m + 64 - j)] * rscale - mx) * rs;
            attn[(base + irow) * 65 + j] = w;
            wb[m * WST + (m + 64 - j)] = (_Float16)w;
        }
        for (int c = 0; c < WST; ++c)
            if (c < m || c > m + 64) wb[m * WST + c] = (_Float16)0.0f;
    }
    __syncthreads();

    // ---- context: 16x128 = W'(16x96) @ Vblk(96x128) ----
    v8f accc[8];
#pragma unroll
    for (int i = 0; i < 8; ++i) accc[i] = zero_v8f();
#pragma unroll
    for (int ks = 0; ks < 3; ++ks) {
        HFrag af;
#pragma unroll
        for (int v = 0; v < 8; ++v)
            af.u[v] = *reinterpret_cast<const unsigned int*>(
                &wb[ln * WST + ks * 32 + a_koff(v, g)]);
#pragma unroll
        for (int half = 0; half < 2; ++half) {
            HFrag bf[4];
#pragma unroll
            for (int j = 0; j < 4; ++j)
#pragma unroll
                for (int v = 0; v < 8; ++v)
                    bf[j].u[v] = *reinterpret_cast<const unsigned int*>(
                        &vt[((half * 4 + j) * 16 + ln) * VST + ks * 32 + g * 16 + 2 * v]);
#pragma unroll
            for (int j = 0; j < 4; ++j)
                accc[half * 4 + j] = __builtin_amdgcn_wmma_f32_16x16x32_f16(
                    false, af.v, false, bf[j].v, (short)0, accc[half * 4 + j], false, false);
        }
    }
    float* cb = reinterpret_cast<float*>(kb);   // reuse K-band LDS
#pragma unroll
    for (int tt = 0; tt < 8; ++tt)
#pragma unroll
        for (int v = 0; v < 8; ++v)
            cb[(v + 8 * g) * 128 + tt * 16 + ln] = accc[tt][v] * rscale;
    __syncthreads();

    // ---- output projection 128 -> 2 ----
    if (lane < 16) {
        int m = lane;
        float o0 = bo[0], o1 = bo[1];
        for (int n = 0; n < 128; ++n) {
            float c = cb[m * 128 + n];
            o0 += c * Wo[n * 2 + 0];
            o1 += c * Wo[n * 2 + 1];
        }
        size_t r = base + i0 + m;
        out[r * 2 + 0] = o0;
        out[r * 2 + 1] = o1;
    }
}

// ---------------------------------------------------------------------------
// Launcher
// ---------------------------------------------------------------------------
extern "C" void kernel_launch(void* const* d_in, const int* in_sizes, int n_in,
                              void* d_out, int out_size, void* d_ws, size_t ws_size,
                              hipStream_t stream)
{
    (void)in_sizes; (void)n_in; (void)out_size; (void)ws_size;
    const float* x  = (const float*)d_in[0];
    const float* W1 = (const float*)d_in[1];
    const float* b1 = (const float*)d_in[2];
    const float* Wq = (const float*)d_in[3];
    const float* Wk = (const float*)d_in[4];
    const float* Wv = (const float*)d_in[5];
    const float* Wo = (const float*)d_in[6];
    const float* bo = (const float*)d_in[7];

    char* p = (char*)d_ws;
    auto take = [&](size_t bytes) {
        char* q = p;
        p += (bytes + 255) & ~(size_t)255;
        return q;
    };
    _Float16* pe  = (_Float16*)take((size_t)S_ * DIN * 2);
    _Float16* w1h = (_Float16*)take((size_t)DIN * H_ * 2);
    _Float16* wqh = (_Float16*)take((size_t)H_ * H_ * 2);
    _Float16* wkh = (_Float16*)take((size_t)H_ * H_ * 2);
    _Float16* wvh = (_Float16*)take((size_t)H_ * H_ * 2);
    _Float16* h16 = (_Float16*)take((size_t)B_ * S_ * H_ * 2);
    _Float16* q16 = (_Float16*)take((size_t)B_ * S_ * H_ * 2);
    _Float16* k16 = (_Float16*)take((size_t)B_ * S_ * H_ * 2);
    _Float16* v16 = (_Float16*)take((size_t)B_ * S_ * H_ * 2);

    float* out  = (float*)d_out;
    float* attn = out + (size_t)B_ * S_ * 2;

    pe_kernel<<<(S_ * DIN) / 256, 256, 0, stream>>>(pe);
    cvt_kernel<<<(DIN * H_ + 255) / 256, 256, 0, stream>>>(W1, w1h, DIN * H_);
    cvt_kernel<<<(H_ * H_ + 255) / 256, 256, 0, stream>>>(Wq, wqh, H_ * H_);
    cvt_kernel<<<(H_ * H_ + 255) / 256, 256, 0, stream>>>(Wk, wkh, H_ * H_);
    cvt_kernel<<<(H_ * H_ + 255) / 256, 256, 0, stream>>>(Wv, wvh, H_ * H_);

    gemm1_kernel<<<(B_ * S_) / 64, 128, 0, stream>>>(x, pe, w1h, b1, h16);
    qkv_kernel<<<dim3((B_ * S_) / 64, 3), 128, 0, stream>>>(h16, wqh, wkh, wvh,
                                                            q16, k16, v16);
    attn_kernel<<<(B_ * S_) / 16, 32, 0, stream>>>(q16, k16, v16, Wo, bo, out, attn);
}
